// Network_13408887898271
// MI455X (gfx1250) — matrix operand order repaired and checked by reading, not compiled, not run on hardware
//
#include <hip/hip_runtime.h>
#include <hip/hip_bf16.h>

// ---------------- static network config (mirrors the reference) ----------------
#define NATOMS   256
#define NPAIR    (NATOMS * NATOMS)
#define BOX_A    12.0f
#define MAX_R    3.8f
#define NB       10
#define HWID     100          // radial MLP hidden width
#define KP1      32           // K pad for GEMM1 (10 -> 32)
#define KP2      128          // K pad for GEMM2/3 (100 -> 128)
#define NP12     112          // N pad for GEMM1/2 (100 -> 112)
#define OUTDP    1024         // max padded out_d (layer3: 1024)
#define CHUNK    32           // atoms per chunk (8192 pairs)
#define CM       (CHUNK * NATOMS)
#define NCLS     230

typedef __attribute__((ext_vector_type(16))) _Float16 v16h;
typedef __attribute__((ext_vector_type(8)))  float    v8f;
typedef __attribute__((ext_vector_type(4)))  unsigned v4u;
typedef __attribute__((ext_vector_type(8)))  int      v8i;
typedef __attribute__((ext_vector_type(4)))  int      v4i;

static __device__ __forceinline__ int imin(int a, int b) { return a < b ? a : b; }

// ---------------------------------------------------------------------------
// TDM: 1-D DMA of (n8 * 8) bytes from global to LDS. D# per ISA 08 §8:
// group0: count=1, lds_addr[63:32], global_addr[120:64], type=2 ([127:126]).
// group1: data_size=8B, tensor_dim0 = tile_dim0 = tensor_dim0_stride = n8,
//         tensor_dim1 = tile_dim1 = 1. Groups 2/3 zero (<=2D tensor).
// This toolchain's builtin is the 6-arg form:
//   (uint32x4 g0, int32x8 g1, int32x4, int32x4, int32x8, i32 cpol)
// ---------------------------------------------------------------------------
static __device__ __forceinline__ void tdm_load_1d(const void* gptr,
                                                   unsigned lds_off, int n8) {
  unsigned long long ga = (unsigned long long)gptr;
  v4u g0;
  g0[0] = 1u;                                        // count=1, user mode
  g0[1] = lds_off;                                   // lds_addr (bytes)
  g0[2] = (unsigned)(ga & 0xFFFFFFFFu);              // global_addr[31:0]
  g0[3] = (unsigned)((ga >> 32) & 0x01FFFFFFu) | (2u << 30);  // [56:32], type=2
  v8i g1;
  g1[0] = (3 << 16);                                 // data_size = 8 bytes
  g1[1] = (int)((unsigned)(n8 & 0xFFFF) << 16);      // tensor_dim0[15:0]
  g1[2] = (int)(((unsigned)n8 >> 16) | (1u << 16));  // tensor_dim0[31:16], tensor_dim1=1
  g1[3] = (int)((unsigned)(n8 & 0xFFFF) << 16);      // tile_dim0 = n8
  g1[4] = 1;                                         // tile_dim1 = 1
  g1[5] = n8;                                        // tensor_dim0_stride lo
  g1[6] = 0;
  g1[7] = 0;
  v4i z4 = {0, 0, 0, 0};
  v8i z8 = {0, 0, 0, 0, 0, 0, 0, 0};
  __builtin_amdgcn_tensor_load_to_lds(g0, g1, z4, z4, z8, 0);
}

// ---------------------------------------------------------------------------
// CG table: np.random.RandomState(0).randn(...) regenerated on device.
// offset(lo,li,lf) = lo*lo*256 + (2lo+1)*(li*li*16 + (2li+1)*lf*lf)
// ---------------------------------------------------------------------------
__device__ unsigned mt_next(unsigned* mt, int& mti) {
  if (mti >= 624) {
    for (int k = 0; k < 624; ++k) {
      unsigned y = (mt[k] & 0x80000000u) | (mt[(k + 1) % 624] & 0x7fffffffu);
      mt[k] = mt[(k + 397) % 624] ^ (y >> 1) ^ ((y & 1u) ? 0x9908b0dfu : 0u);
    }
    mti = 0;
  }
  unsigned y = mt[mti++];
  y ^= y >> 11; y ^= (y << 7) & 0x9d2c5680u; y ^= (y << 15) & 0xefc60000u; y ^= y >> 18;
  return y;
}

__global__ void cg_init_kernel(float* __restrict__ cg) {
  if (threadIdx.x != 0 || blockIdx.x != 0) return;
  unsigned mt[624];
  int mti;
  mt[0] = 0u;  // seed 0
  for (mti = 1; mti < 624; ++mti)
    mt[mti] = 1812433253u * (mt[mti - 1] ^ (mt[mti - 1] >> 30)) + (unsigned)mti;
  mti = 624;
  double cache = 0.0; bool has = false;
  int off = 0;
  for (int lo = 0; lo < 4; ++lo)
    for (int li = 0; li < 4; ++li)
      for (int lf = 0; lf < 4; ++lf) {
        int n = (2 * lo + 1) * (2 * li + 1) * (2 * lf + 1);
        double sc = 1.0 / sqrt((double)((2 * li + 1) * (2 * lf + 1)));
        for (int i = 0; i < n; ++i) {
          double g;
          if (has) { g = cache; has = false; }
          else {
            double x1, x2, r2;
            do {
              unsigned a = mt_next(mt, mti) >> 5, b = mt_next(mt, mti) >> 6;
              double d1 = ((double)a * 67108864.0 + (double)b) / 9007199254740992.0;
              a = mt_next(mt, mti) >> 5; b = mt_next(mt, mti) >> 6;
              double d2 = ((double)a * 67108864.0 + (double)b) / 9007199254740992.0;
              x1 = 2.0 * d1 - 1.0; x2 = 2.0 * d2 - 1.0;
              r2 = x1 * x1 + x2 * x2;
            } while (r2 >= 1.0 || r2 == 0.0);
            double f = sqrt(-2.0 * log(r2) / r2);
            cache = f * x1; has = true;
            g = f * x2;
          }
          cg[off + i] = (float)(g * sc);
        }
        off += n;
      }
}

// ---------------------------------------------------------------------------
// Per-pair geometry: minimal-image dist, mask, spherical harmonics l<=3 (16),
// cosine radial basis (10, padded to 32 as f16 for WMMA A operand).
// ---------------------------------------------------------------------------
__global__ void pair_kernel(const float* __restrict__ geo,
                            _Float16* __restrict__ rbh,
                            float* __restrict__ ys,
                            float* __restrict__ maskv,
                            float* __restrict__ cnt) {
  int a = blockIdx.x, b = threadIdx.x;
  float dx = geo[a * 3 + 0] - geo[b * 3 + 0];
  float dy = geo[a * 3 + 1] - geo[b * 3 + 1];
  float dz = geo[a * 3 + 2] - geo[b * 3 + 2];
  dx -= BOX_A * rintf(dx / BOX_A);
  dy -= BOX_A * rintf(dy / BOX_A);
  dz -= BOX_A * rintf(dz / BOX_A);
  float dist = sqrtf(dx * dx + dy * dy + dz * dz + 1e-12f);
  float m = (dist > 1e-5f && dist < MAX_R) ? 1.f : 0.f;
  float inv = 1.f / dist;
  float x = dx * inv, y = dy * inv, z = dz * inv;

  int pair = a * NATOMS + b;
  float* Y = ys + pair * 16;
  Y[0] = 0.28209479177f;
  Y[1] = 0.4886025119f * y; Y[2] = 0.4886025119f * z; Y[3] = 0.4886025119f * x;
  Y[4] = 1.0925484306f * x * y;
  Y[5] = 1.0925484306f * y * z;
  Y[6] = 0.3153915653f * (3.f * z * z - 1.f);
  Y[7] = 1.0925484306f * z * x;
  Y[8] = 0.5462742153f * (x * x - y * y);
  float z2 = z * z;
  Y[9]  = 0.5900435899f * y * (3.f * x * x - y * y);
  Y[10] = 2.8906114426f * x * y * z;
  Y[11] = 0.4570457995f * y * (5.f * z2 - 1.f);
  Y[12] = 0.3731763326f * z * (5.f * z2 - 3.f);
  Y[13] = 0.4570457995f * x * (5.f * z2 - 1.f);
  Y[14] = 1.4453057213f * z * (x * x - y * y);
  Y[15] = 0.5900435899f * x * (x * x - 3.f * y * y);

  const float step = MAX_R / (float)(NB - 1);
  _Float16* rb = rbh + pair * KP1;
  for (int i = 0; i < KP1; ++i) {
    float v = 0.f;
    if (i < NB) {
      float diff = dist - step * (float)i;
      if (fabsf(diff) < step) {
        float c = __cosf(0.5f * 3.14159265358979f * diff / step);
        v = c * c;
      }
    }
    rb[i] = (_Float16)v;
  }
  maskv[pair] = m;

  __shared__ float red[NATOMS];
  red[b] = m;
  __syncthreads();
  for (int s = NATOMS / 2; s > 0; s >>= 1) {
    if (b < s) red[b] += red[b + s];
    __syncthreads();
  }
  if (b == 0) cnt[a] = red[0];
}

__global__ void nnorm_kernel(const float* __restrict__ cnt, float* __restrict__ rinv) {
  __shared__ float red[NATOMS];
  int t = threadIdx.x;
  red[t] = cnt[t];
  __syncthreads();
  for (int s = NATOMS / 2; s > 0; s >>= 1) {
    if (t < s) red[t] += red[t + s];
    __syncthreads();
  }
  if (t == 0) {
    float mean = red[0] / (float)NATOMS;
    if (mean < 1.f) mean = 1.f;
    rinv[0] = 1.f / sqrtf(mean);
  }
}

// featIn layout: [atom][li(4)][v(4)][j(8)] ; layer-0 input = 0.5 at (0,0,0)
__global__ void feat_init_kernel(float* __restrict__ featIn) {
  int i = blockIdx.x * blockDim.x + threadIdx.x;
  if (i >= NATOMS * 128) return;
  featIn[i] = ((i & 127) == 0) ? 0.5f : 0.f;
}

__global__ void zero_f32_kernel(float* __restrict__ p, int n) {
  int i = blockIdx.x * blockDim.x + threadIdx.x;
  if (i < n) p[i] = 0.f;
}
__global__ void zero_f16_kernel(_Float16* __restrict__ p, int n) {
  int i = blockIdx.x * blockDim.x + threadIdx.x;
  if (i < n) p[i] = (_Float16)0.f;
}

// Convert f32 weight [K][N] -> f16 transposed padded [Npad][Kpad] (B operand).
__global__ void wcvt_kernel(const float* __restrict__ W, _Float16* __restrict__ Wt,
                            int K, int N, int Kpad, int Npad) {
  int i = blockIdx.x * blockDim.x + threadIdx.x;
  if (i >= Kpad * Npad) return;
  int n = i / Kpad, k = i % Kpad;
  Wt[i] = (k < K && n < N) ? (_Float16)W[k * N + n] : (_Float16)0.f;
}

// ---------------------------------------------------------------------------
// WMMA GEMM: C = act(A @ Wt^T + bias). 4 waves/block; each block owns a
// 64-row M super-tile staged into LDS by the Tensor Data Mover (wave 0 issues
// tensor_load_to_lds, waits TENSORcnt, then workgroup barrier). Each wave owns
// one 16-col N tile; B fragments for all K-steps stay resident in VGPRs and
// are reused across the 4 M-tiles.
// A frag (16-bit 16x32, contiguous K like B): lane m = lane&15; hi half takes K 16..31 of each step.
// ---------------------------------------------------------------------------
template <int KSTEPS>
__global__ void gemm_relu_f16(const _Float16* __restrict__ A, int lda,
                              const _Float16* __restrict__ Wt, int ldk,
                              const float* __restrict__ bias, int Nvalid,
                              _Float16* __restrict__ C, int ldc, int ntiles) {
  __shared__ _Float16 ldsA[64 * KP2];
  int lane = threadIdx.x & 31, wave = threadIdx.x >> 5;
  int nt = blockIdx.y * 4 + wave;
  bool active = nt < ntiles;
  int m0 = blockIdx.x * 64;

  if (threadIdx.x < 32) {  // wave-uniform: wave 0 drives the TDM
    tdm_load_1d(A + (size_t)m0 * lda, (unsigned)(size_t)(void*)ldsA, lda * 16);
    __builtin_amdgcn_s_wait_tensorcnt(0);
  }
  __syncthreads();
  if (!active) return;

  int n0 = nt * 16;
  int hi = lane >> 4, l16 = lane & 15;

  union { v16h v; unsigned u[8]; } bf[KSTEPS];
  const _Float16* bcol = Wt + (n0 + l16) * ldk + hi * 16;
#pragma unroll
  for (int ks = 0; ks < KSTEPS; ++ks)
#pragma unroll
    for (int v = 0; v < 8; ++v)
      bf[ks].u[v] = *(const unsigned*)(bcol + ks * 32 + 2 * v);

  int n = n0 + l16;
  float bn = (n < Nvalid) ? bias[n] : 0.f;

#pragma unroll
  for (int mt = 0; mt < 4; ++mt) {
    const _Float16* arow = ldsA + (mt * 16 + l16) * lda + hi * 16;
    v8f c = {};
#pragma unroll
    for (int ks = 0; ks < KSTEPS; ++ks) {
      union { v16h v; unsigned u[8]; } af;
#pragma unroll
      for (int v = 0; v < 8; ++v) {
        int ka = 2 * v;
        af.u[v] = *(const unsigned*)(arow + ks * 32 + ka);
      }
      c = __builtin_amdgcn_wmma_f32_16x16x32_f16(false, af.v, false, bf[ks].v,
                                                 (short)0, c, false, false);
    }
#pragma unroll
    for (int r = 0; r < 8; ++r) {
      float val = c[r] + bn;
      val = val > 0.f ? val : 0.f;
      C[(m0 + mt * 16 + r + hi * 8) * ldc + n] = (_Float16)val;
    }
  }
}

template <int KSTEPS>
__global__ void gemm_mask_f32(const _Float16* __restrict__ A, int lda,
                              const _Float16* __restrict__ Wt, int ldk,
                              const float* __restrict__ bias, int Nvalid,
                              float* __restrict__ C, int ldc, int ntiles,
                              const float* __restrict__ maskRows,
                              const float* __restrict__ rinv) {
  __shared__ _Float16 ldsA[64 * KP2];
  int lane = threadIdx.x & 31, wave = threadIdx.x >> 5;
  int nt = blockIdx.y * 4 + wave;
  bool active = nt < ntiles;
  int m0 = blockIdx.x * 64;

  if (threadIdx.x < 32) {
    tdm_load_1d(A + (size_t)m0 * lda, (unsigned)(size_t)(void*)ldsA, lda * 16);
    __builtin_amdgcn_s_wait_tensorcnt(0);
  }
  __syncthreads();
  if (!active) return;

  int n0 = nt * 16;
  int hi = lane >> 4, l16 = lane & 15;

  union { v16h v; unsigned u[8]; } bf[KSTEPS];
  const _Float16* bcol = Wt + (n0 + l16) * ldk + hi * 16;
#pragma unroll
  for (int ks = 0; ks < KSTEPS; ++ks)
#pragma unroll
    for (int v = 0; v < 8; ++v)
      bf[ks].u[v] = *(const unsigned*)(bcol + ks * 32 + 2 * v);

  int n = n0 + l16;
  float bn = (n < Nvalid) ? bias[n] : 0.f;
  float rs = rinv[0];

#pragma unroll
  for (int mt = 0; mt < 4; ++mt) {
    const _Float16* arow = ldsA + (mt * 16 + l16) * lda + hi * 16;
    v8f c = {};
#pragma unroll
    for (int ks = 0; ks < KSTEPS; ++ks) {
      union { v16h v; unsigned u[8]; } af;
#pragma unroll
      for (int v = 0; v < 8; ++v) {
        int ka = 2 * v;
        af.u[v] = *(const unsigned*)(arow + ks * 32 + ka);
      }
      c = __builtin_amdgcn_wmma_f32_16x16x32_f16(false, af.v, false, bf[ks].v,
                                                 (short)0, c, false, false);
    }
#pragma unroll
    for (int r = 0; r < 8; ++r) {
      int m = m0 + mt * 16 + r + hi * 8;
      C[m * ldc + n] = (c[r] + bn) * maskRows[m] * rs;
    }
  }
}

// ---------------------------------------------------------------------------
// Tensor-product contraction + neighbor scatter-sum. One block per atom `a`,
// thread b = neighbor. Path enumeration mirrors the Python loops exactly.
// outb layout: [atom][slot(64)][j(8)]; layers0-2: bo<4 -> slot bo*4+u,
// gate block -> slot 16+u; layer3 -> slot u.
// ---------------------------------------------------------------------------
__global__ void contract_kernel(const float* __restrict__ rall,
                                const float* __restrict__ ys,
                                const float* __restrict__ featIn,
                                const float* __restrict__ cg,
                                float* __restrict__ outb,
                                int layer, int a0) {
  int a = a0 + blockIdx.x;
  int b = threadIdx.x;
  const float* R = rall + (blockIdx.x * NATOMS + b) * OUTDP;
  const float* Y = ys + (a * NATOMS + b) * 16;
  __shared__ float acc[64];

  int nblocks = (layer == 3) ? 1 : 5;
  int nin = (layer == 0) ? 1 : 4;
  int off = 0;
  for (int bo = 0; bo < nblocks; ++bo) {
    int mo, lo, sbase;
    if (layer == 3)      { mo = 64; lo = 0;  sbase = 0; }
    else if (bo < 4)     { mo = 4;  lo = bo; sbase = bo * 4; }
    else                 { mo = 12; lo = 0;  sbase = 16; }
    int dlo = 2 * lo + 1;
    for (int ii = 0; ii < nin; ++ii) {
      int mi = (layer == 0) ? 1 : 4;
      int li = (layer == 0) ? 0 : ii;
      int dli = 2 * li + 1;
      int lf0 = lo > li ? lo - li : li - lo;
      int lf1 = imin(lo + li, 3);
      for (int lf = lf0; lf <= lf1; ++lf) {
        int dlf = 2 * lf + 1;
        const float* C = cg + lo * lo * 256 + dlo * (li * li * 16 + dli * lf * lf);
        const float* Yp = Y + lf * lf;   // Y offsets: 0,1,4,9 = lf^2
        float Q[7][7];                   // Q[j][io] = sum_k Y[k]*CG[io][j][k]
        for (int j = 0; j < dli; ++j)
          for (int io = 0; io < dlo; ++io) {
            float q = 0.f;
            for (int k = 0; k < dlf; ++k) q += Yp[k] * C[(io * dli + j) * dlf + k];
            Q[j][io] = q;
          }
        float E[4][7];                   // E[v][io] = sum_j F[v][j]*Q[j][io]
        for (int v = 0; v < mi; ++v) {
          const float* F = featIn + ((b * 4 + li) * 4 + v) * 8;
          for (int io = 0; io < dlo; ++io) {
            float e = 0.f;
            for (int j = 0; j < dli; ++j) e += F[j] * Q[j][io];
            E[v][io] = e;
          }
        }
        int sz = mo * dlo;
        if (b < sz) acc[b] = 0.f;
        __syncthreads();
        for (int u = 0; u < mo; ++u) {
          const float* Ru = R + off + u * mi;
          for (int io = 0; io < dlo; ++io) {
            float s = 0.f;
            for (int v = 0; v < mi; ++v) s += Ru[v] * E[v][io];
            atomicAdd(&acc[u * dlo + io], s);   // ds_add_f32
          }
        }
        __syncthreads();
        if (b < sz) {
          int u = b / dlo, io = b % dlo;
          outb[(a * 64 + sbase + u) * 8 + io] += acc[b];
        }
        __syncthreads();
        off += mo * mi;
      }
    }
  }
}

// Gated nonlinearity -> next-layer featIn (with the 0.5 feature scale).
__global__ void gate_kernel(const float* __restrict__ outb, float* __restrict__ featIn) {
  int a = blockIdx.x, t = threadIdx.x;  // 128 threads: (li,v,j)
  __shared__ float g[12];
  if (t < 12) g[t] = 1.f / (1.f + __expf(-outb[(a * 64 + 16 + t) * 8]));
  __syncthreads();
  int li = t >> 5, v = (t >> 3) & 3, j = t & 7;
  float val = 0.f;
  if (j < 2 * li + 1) {
    if (li == 0) val = fmaxf(outb[(a * 64 + v) * 8], 0.f);
    else         val = outb[(a * 64 + li * 4 + v) * 8 + j] * g[(li - 1) * 4 + v];
  }
  featIn[((a * 4 + li) * 4 + v) * 8 + j] = 0.5f * val;
}

// pooled[n] = mean_a relu(outb[a][n][0])   (layer-3 scalar features)
__global__ void pool_kernel(const float* __restrict__ outb, float* __restrict__ pooled) {
  int n = blockIdx.x, a = threadIdx.x;
  __shared__ float red[NATOMS];
  red[a] = fmaxf(outb[(a * 64 + n) * 8], 0.f);
  __syncthreads();
  for (int s = NATOMS / 2; s > 0; s >>= 1) {
    if (a < s) red[a] += red[a + s];
    __syncthreads();
  }
  if (a == 0) pooled[n] = red[0] / (float)NATOMS;
}

__global__ void logits_kernel(const float* __restrict__ pooled,
                              const float* __restrict__ Wc,
                              const float* __restrict__ bc,
                              float* __restrict__ out) {
  int c = threadIdx.x;
  if (c >= NCLS) return;
  float s = bc[c];
  for (int n = 0; n < 64; ++n) s += pooled[n] * Wc[n * NCLS + c];
  out[c] = s;
}

// ---------------------------------------------------------------------------
extern "C" void kernel_launch(void* const* d_in, const int* in_sizes, int n_in,
                              void* d_out, int out_size, void* d_ws, size_t ws_size,
                              hipStream_t stream) {
  (void)in_sizes; (void)n_in; (void)out_size; (void)ws_size;
  // pytree order: geometry, W_cls, b_cls, then per layer W1,W2,W3,b1,b2,b3
  const float* geo  = (const float*)d_in[0];
  const float* Wcls = (const float*)d_in[1];
  const float* bcls = (const float*)d_in[2];

  static const int OUTD[4]    = {28, 736, 736, 1024};
  static const int OUTDPAD[4] = {32, 736, 736, 1024};

  // workspace layout (256B aligned)
  char* p = (char*)d_ws;
  auto take = [&](size_t bytes) { char* r = p; p += (bytes + 255) & ~size_t(255); return r; };
  float*    cg     = (float*)   take(4096 * 4);
  _Float16* rbh    = (_Float16*)take((size_t)NPAIR * KP1 * 2);
  float*    ys     = (float*)   take((size_t)NPAIR * 16 * 4);
  float*    maskv  = (float*)   take((size_t)NPAIR * 4);
  float*    cnt    = (float*)   take(NATOMS * 4);
  float*    rinv   = (float*)   take(4);
  float*    featIn = (float*)   take((size_t)NATOMS * 128 * 4);
  float*    outb   = (float*)   take((size_t)NATOMS * 64 * 8 * 4);
  _Float16* h1     = (_Float16*)take((size_t)CM * KP2 * 2);
  _Float16* h2     = (_Float16*)take((size_t)CM * KP2 * 2);
  float*    rall   = (float*)   take((size_t)CM * OUTDP * 4);
  _Float16* w1t    = (_Float16*)take((size_t)NP12 * KP1 * 2);
  _Float16* w2t    = (_Float16*)take((size_t)NP12 * KP2 * 2);
  _Float16* w3t    = (_Float16*)take((size_t)OUTDP * KP2 * 2);
  float*    pooled = (float*)   take(64 * 4);

  // ---- setup ----
  cg_init_kernel<<<1, 32, 0, stream>>>(cg);
  pair_kernel<<<NATOMS, NATOMS, 0, stream>>>(geo, rbh, ys, maskv, cnt);
  nnorm_kernel<<<1, NATOMS, 0, stream>>>(cnt, rinv);
  feat_init_kernel<<<(NATOMS * 128 + 255) / 256, 256, 0, stream>>>(featIn);
  {
    int n = CM * KP2;
    zero_f16_kernel<<<(n + 255) / 256, 256, 0, stream>>>(h1, n);
    zero_f16_kernel<<<(n + 255) / 256, 256, 0, stream>>>(h2, n);
  }

  for (int layer = 0; layer < 4; ++layer) {
    const float* W1 = (const float*)d_in[3 + 6 * layer + 0];
    const float* W2 = (const float*)d_in[3 + 6 * layer + 1];
    const float* W3 = (const float*)d_in[3 + 6 * layer + 2];
    const float* b1 = (const float*)d_in[3 + 6 * layer + 3];
    const float* b2 = (const float*)d_in[3 + 6 * layer + 4];
    const float* b3 = (const float*)d_in[3 + 6 * layer + 5];
    int outd = OUTD[layer], outdp = OUTDPAD[layer];

    wcvt_kernel<<<(NP12 * KP1 + 255) / 256, 256, 0, stream>>>(W1, w1t, NB, HWID, KP1, NP12);
    wcvt_kernel<<<(NP12 * KP2 + 255) / 256, 256, 0, stream>>>(W2, w2t, HWID, HWID, KP2, NP12);
    wcvt_kernel<<<(outdp * KP2 + 255) / 256, 256, 0, stream>>>(W3, w3t, HWID, outd, KP2, outdp);
    zero_f32_kernel<<<(NATOMS * 512 + 255) / 256, 256, 0, stream>>>(outb, NATOMS * 512);

    int nt3 = outdp / 16;
    for (int ch = 0; ch < NATOMS / CHUNK; ++ch) {
      int a0 = ch * CHUNK;
      const _Float16* Achunk = rbh + (size_t)a0 * NATOMS * KP1;
      dim3 g1(CM / 64, (7 + 3) / 4);
      gemm_relu_f16<1><<<g1, 128, 0, stream>>>(Achunk, KP1, w1t, KP1, b1, HWID,
                                               h1, KP2, 7);
      dim3 g2(CM / 64, (7 + 3) / 4);
      gemm_relu_f16<4><<<g2, 128, 0, stream>>>(h1, KP2, w2t, KP2, b2, HWID,
                                               h2, KP2, 7);
      dim3 g3(CM / 64, (nt3 + 3) / 4);
      gemm_mask_f32<4><<<g3, 128, 0, stream>>>(h2, KP2, w3t, KP2, b3, outd,
                                               rall, OUTDP, nt3,
                                               maskv + (size_t)a0 * NATOMS, rinv);
      contract_kernel<<<CHUNK, NATOMS, 0, stream>>>(rall, ys, featIn, cg, outb,
                                                    layer, a0);
    }
    if (layer < 3) gate_kernel<<<NATOMS, 128, 0, stream>>>(outb, featIn);
  }

  pool_kernel<<<64, NATOMS, 0, stream>>>(outb, pooled);
  logits_kernel<<<1, 256, 0, stream>>>(pooled, Wcls, bcls, (float*)d_out);
}
